// Rvt_80195629351345
// MI455X (gfx1250) — matrix operand order repaired and checked by reading, not compiled, not run on hardware
//
#include <hip/hip_runtime.h>
#include <hip/hip_bf16.h>
#include <math.h>
#include <stdint.h>

// ---- dims -------------------------------------------------------------
#define NB   8
#define CH   128
#define HH   36
#define WW   36
#define QTOK 1296      // 36*36
#define DTOK 1156      // 34*34
#define DPAD 1184      // 37*32, padded for 32-wide flash steps
#define KCV  1152      // 128*9 implicit-GEMM K for 3x3 convs

typedef __attribute__((ext_vector_type(16))) __bf16         v16bf;
typedef __attribute__((ext_vector_type(8)))  float          v8f;
typedef __attribute__((ext_vector_type(16))) unsigned short v16u;
typedef __attribute__((ext_vector_type(8)))  unsigned short v8u;

static __device__ __forceinline__ unsigned short f2bf(float f) {
    union { float f; unsigned int u; } c; c.f = f;
    unsigned int u = c.u;
    return (unsigned short)((u + 0x7FFFu + ((u >> 16) & 1u)) >> 16);
}
static __device__ __forceinline__ v8f zero8() {
    v8f z;
#pragma unroll
    for (int i = 0; i < 8; ++i) z[i] = 0.f;
    return z;
}
static __device__ __forceinline__ v16u zero16u() {
    v16u z;
#pragma unroll
    for (int i = 0; i < 16; ++i) z[i] = 0;
    return z;
}

// A fragment (16xK tile, 16-bit): lane&15 = row M; lanes 0-15 hold K 0..15,
// lanes 16-31 hold K 16..31 (contiguous, matching loadB).
static __device__ __forceinline__ v16bf loadA(const unsigned short* p0, int ld, int lane) {
    const unsigned short* p = p0 + (size_t)(lane & 15) * ld + ((lane >> 4) ? 16 : 0);
    return __builtin_bit_cast(v16bf, *(const v16u*)p);
}
// B fragment (Kx16 tile, 16-bit): lane&15 = column N; lanes 0-15 hold K 0..15,
// lanes 16-31 hold K 16..31. Source is K-contiguous per column.
static __device__ __forceinline__ v16bf loadB(const unsigned short* p0, int ld, int lane) {
    const unsigned short* p = p0 + (size_t)(lane & 15) * ld + ((lane >> 4) ? 16 : 0);
    return __builtin_bit_cast(v16bf, *(const v16u*)p);
}
static __device__ __forceinline__ v8f wmma_bf16(v16bf a, v16bf b, v8f c) {
    return __builtin_amdgcn_wmma_f32_16x16x32_bf16(false, a, false, b, (short)0, c, false, false);
}
// async copy of 16 bytes per lane: global -> LDS (ASYNCcnt-tracked)
static __device__ __forceinline__ void async_b128(unsigned lds_off, const void* gaddr) {
    asm volatile("global_load_async_to_lds_b128 %0, %1, off" :: "v"(lds_off), "v"(gaddr) : "memory");
}

// ---- prep kernels -----------------------------------------------------
__global__ void cast_bf16_k(unsigned short* dst, const float* src, int n) {
    int i = blockIdx.x * blockDim.x + threadIdx.x;
    if (i < n) dst[i] = f2bf(src[i]);
}
// OIHW [co][ci][3][3] -> [co][tap][ci] so WMMA K-chunks stay within one tap
__global__ void reorder_w_k(unsigned short* dst, const float* src) {
    int i = blockIdx.x * blockDim.x + threadIdx.x;
    if (i >= CH * KCV) return;
    int ci = i & 127;
    int tap = (i >> 7) % 9;
    int co = i / KCV;
    dst[i] = f2bf(src[((size_t)co * CH + ci) * 9 + tap]);
}
// NCHW f32 -> token-major [n][hw][ci] bf16
__global__ void transpose_inp_k(unsigned short* xt, const float* inp) {
    int i = blockIdx.x * blockDim.x + threadIdx.x;
    if (i >= NB * CH * QTOK) return;
    int ci = i & 127;
    int rest = i >> 7;
    int hw = rest % QTOK;
    int n = rest / QTOK;
    xt[i] = f2bf(inp[((size_t)n * CH + ci) * QTOK + hw]);
}
// v_c[co] = sum_ci (sum_9 w_vc[co][ci][.]) * c0[ci]
__global__ void vc_k(float* vc, unsigned short* vcb, const float* wvc, const float* c0) {
    int co = threadIdx.x;
    float s = 0.f;
    for (int ci = 0; ci < CH; ++ci) {
        const float* p = wvc + ((size_t)co * CH + ci) * 9;
        float w9 = 0.f;
#pragma unroll
        for (int t = 0; t < 9; ++t) w9 += p[t];
        s += w9 * c0[ci];
    }
    vc[co] = s;
    vcb[co] = f2bf(s);
}

// ---- proj_x: x = W_x * inp + b_x, store NHWC bf16 ---------------------
__global__ __launch_bounds__(32) void proj_x_k(const unsigned short* __restrict__ XT,
                                               const unsigned short* __restrict__ WX,
                                               const float* __restrict__ BX,
                                               unsigned short* __restrict__ X) {
    int lane = threadIdx.x, n = blockIdx.y, t0 = blockIdx.x * 16;
    int col = lane & 15, hi = lane >> 4, rb = hi * 8;
    int tok = t0 + col;
    v16bf bfr[4];
#pragma unroll
    for (int j = 0; j < 4; ++j)
        bfr[j] = loadB(XT + ((size_t)(n * QTOK + t0)) * CH + j * 32, CH, lane);
#pragma unroll
    for (int i = 0; i < 8; ++i) {
        v16bf a[4];
#pragma unroll
        for (int j = 0; j < 4; ++j) a[j] = loadA(WX + (size_t)(i * 16) * CH + j * 32, CH, lane);
        v8f acc = zero8();
#pragma unroll
        for (int j = 0; j < 4; ++j) acc = wmma_bf16(a[j], bfr[j], acc);
        v8u pk;
#pragma unroll
        for (int r = 0; r < 8; ++r) pk[r] = f2bf(acc[r] + BX[i * 16 + rb + r]);
        *(v8u*)(X + ((size_t)(n * QTOK + tok)) * CH + i * 16 + rb) = pk;
    }
}

// ---- 3x3 convs as implicit GEMM with async-LDS weight staging ---------
// Weight chunk for K-step kc: rows co=0..127, 32 halves each (8 KB), double buffered.
static __device__ __forceinline__ void stage_w(const unsigned short* W, int kc,
                                               unsigned lbase, int lane) {
    int rowb = lane >> 2;   // 0..7
    int part = lane & 3;    // 0..3
    const unsigned short* g0 = W + (size_t)rowb * KCV + kc * 32 + part * 8;
    unsigned l0 = lbase + (unsigned)(rowb * 32 + part * 8) * 2;
#pragma unroll
    for (int t = 0; t < 16; ++t)
        async_b128(l0 + (unsigned)t * 512, (const void*)(g0 + (size_t)t * 8 * KCV));
}

// q conv (3x3 SAME), out token-major [n][q][co]
__global__ __launch_bounds__(32) void conv_q_k(const unsigned short* __restrict__ X,
                                               const unsigned short* __restrict__ W,
                                               unsigned short* __restrict__ Qo) {
    int lane = threadIdx.x, n = blockIdx.y, t0 = blockIdx.x * 16;
    int col = lane & 15, hi = lane >> 4, rb = hi * 8;
    int tok = t0 + col;
    int oy = tok / WW, ox = tok % WW;
    __shared__ unsigned short Wl[2][CH * 32];
    unsigned lb = (unsigned)(uintptr_t)&Wl[0][0];
    v8f acc[8];
#pragma unroll
    for (int i = 0; i < 8; ++i) acc[i] = zero8();
    stage_w(W, 0, lb, lane);
    for (int kc = 0; kc < 36; ++kc) {
        if (kc + 1 < 36) {
            stage_w(W, kc + 1, lb + (unsigned)(((kc + 1) & 1) * CH * 32 * 2), lane);
            asm volatile("s_wait_asynccnt 0x10" ::: "memory");
        } else {
            asm volatile("s_wait_asynccnt 0x0" ::: "memory");
        }
        int kk = kc * 32 + hi * 16;
        int tap = kk >> 7, ci0 = kk & 127;
        int py = oy + tap / 3 - 1, px = ox + tap % 3 - 1;
        v16u braw = zero16u();
        if ((unsigned)py < HH && (unsigned)px < WW)
            braw = *(const v16u*)(X + ((size_t)(n * QTOK + py * WW + px)) * CH + ci0);
        v16bf b = __builtin_bit_cast(v16bf, braw);
        const unsigned short* Wc = &Wl[kc & 1][0];
        v16bf a[8];
#pragma unroll
        for (int i = 0; i < 8; ++i) a[i] = loadA(Wc + i * 16 * 32, 32, lane);
#pragma unroll
        for (int i = 0; i < 8; ++i) acc[i] = wmma_bf16(a[i], b, acc[i]);
    }
#pragma unroll
    for (int i = 0; i < 8; ++i) {
        v8u pk;
#pragma unroll
        for (int r = 0; r < 8; ++r) pk[r] = f2bf(acc[i][r]);
        *(v8u*)(Qo + ((size_t)(n * QTOK + tok)) * CH + i * 16 + rb) = pk;
    }
}

// k/v conv (3x3 VALID), zero-padded to DPAD tokens.
// chanMajor=0: token-major [n][d][co] (K).  chanMajor=1: [n][co][DPAD] (V).
__global__ __launch_bounds__(32) void conv_kv_k(const unsigned short* __restrict__ X,
                                                const unsigned short* __restrict__ W,
                                                unsigned short* __restrict__ Out,
                                                int chanMajor) {
    int lane = threadIdx.x, n = blockIdx.y, t0 = blockIdx.x * 16;
    int col = lane & 15, hi = lane >> 4, rb = hi * 8;
    int tok = t0 + col;
    int valid = tok < DTOK;
    int oy = valid ? tok / 34 : 0, ox = valid ? tok % 34 : 0;
    __shared__ unsigned short Wl[2][CH * 32];
    unsigned lb = (unsigned)(uintptr_t)&Wl[0][0];
    v8f acc[8];
#pragma unroll
    for (int i = 0; i < 8; ++i) acc[i] = zero8();
    stage_w(W, 0, lb, lane);
    for (int kc = 0; kc < 36; ++kc) {
        if (kc + 1 < 36) {
            stage_w(W, kc + 1, lb + (unsigned)(((kc + 1) & 1) * CH * 32 * 2), lane);
            asm volatile("s_wait_asynccnt 0x10" ::: "memory");
        } else {
            asm volatile("s_wait_asynccnt 0x0" ::: "memory");
        }
        int kk = kc * 32 + hi * 16;
        int tap = kk >> 7, ci0 = kk & 127;
        int py = oy + tap / 3, px = ox + tap % 3;
        v16u braw = zero16u();
        if (valid)
            braw = *(const v16u*)(X + ((size_t)(n * QTOK + py * WW + px)) * CH + ci0);
        v16bf b = __builtin_bit_cast(v16bf, braw);
        const unsigned short* Wc = &Wl[kc & 1][0];
        v16bf a[8];
#pragma unroll
        for (int i = 0; i < 8; ++i) a[i] = loadA(Wc + i * 16 * 32, 32, lane);
#pragma unroll
        for (int i = 0; i < 8; ++i) acc[i] = wmma_bf16(a[i], b, acc[i]);
    }
    if (chanMajor) {
#pragma unroll
        for (int i = 0; i < 8; ++i)
#pragma unroll
            for (int r = 0; r < 8; ++r)
                Out[((size_t)(n * CH + i * 16 + rb + r)) * DPAD + tok] = f2bf(acc[i][r]);
    } else {
#pragma unroll
        for (int i = 0; i < 8; ++i) {
            v8u pk;
#pragma unroll
            for (int r = 0; r < 8; ++r) pk[r] = f2bf(acc[i][r]);
            *(v8u*)(Out + ((size_t)(n * DPAD + tok)) * CH + i * 16 + rb) = pk;
        }
    }
}

// ---- flash attention: a0 = softmax(q_x^T k_x) @ v_x -------------------
__global__ __launch_bounds__(32) void flash_k(const unsigned short* __restrict__ Q,
                                              const unsigned short* __restrict__ K,
                                              const unsigned short* __restrict__ V,
                                              unsigned short* __restrict__ A0) {
    int lane = threadIdx.x, n = blockIdx.y, q0 = blockIdx.x * 16;
    int col = lane & 15, hi = lane >> 4, rb = hi * 8;
    __shared__ unsigned short KL[2][32 * CH];   // [d][ci], 8 KB each
    __shared__ unsigned short VL[2][CH * 32];   // [co][d], 8 KB each
    __shared__ unsigned short P[16 * 32];
    unsigned kb0 = (unsigned)(uintptr_t)&KL[0][0];
    unsigned vb0 = (unsigned)(uintptr_t)&VL[0][0];

    v16bf qa[4];
#pragma unroll
    for (int j = 0; j < 4; ++j)
        qa[j] = loadA(Q + ((size_t)(n * QTOK + q0)) * CH + j * 32, CH, lane);

    v8f o[8];
#pragma unroll
    for (int i = 0; i < 8; ++i) o[i] = zero8();
    float m[8], l[8];
#pragma unroll
    for (int r = 0; r < 8; ++r) { m[r] = -3.0e38f; l[r] = 0.f; }

    // stage K (global stride == LDS stride == 256B/row -> shared inst offset)
    // and V (strided rows) for tile `it` into buffer `buf`: 32 async ops.
    auto stage = [&](int it, int buf) {
        int d0 = it * 32;
        {
            int row2 = lane >> 4, part = lane & 15;
            const void* g = (const void*)(K + ((size_t)(n * DPAD + d0 + row2)) * CH + part * 8);
            unsigned l0 = kb0 + (unsigned)buf * (32 * CH * 2) + (unsigned)(row2 * CH + part * 8) * 2;
#pragma unroll
            for (int t = 0; t < 16; ++t)
                asm volatile("global_load_async_to_lds_b128 %0, %1, off offset:%2"
                             :: "v"(l0), "v"(g), "i"(t * 512) : "memory");
        }
        {
            int rowb = lane >> 2, part = lane & 3;
            const unsigned short* g0 = V + ((size_t)(n * CH + rowb)) * DPAD + d0 + part * 8;
            unsigned l0 = vb0 + (unsigned)buf * (CH * 32 * 2) + (unsigned)(rowb * 32 + part * 8) * 2;
#pragma unroll
            for (int t = 0; t < 16; ++t)
                async_b128(l0 + (unsigned)t * 512, (const void*)(g0 + (size_t)t * 8 * DPAD));
        }
    };

    stage(0, 0);
    for (int it = 0; it < DPAD / 32; ++it) {
        int d0 = it * 32;
        if (it + 1 < DPAD / 32) {
            stage(it + 1, (it + 1) & 1);
            asm volatile("s_wait_asynccnt 0x20" ::: "memory");
        } else {
            asm volatile("s_wait_asynccnt 0x0" ::: "memory");
        }
        const unsigned short* Kc = &KL[it & 1][0];
        const unsigned short* Vc = &VL[it & 1][0];

        v8f s0 = zero8(), s1 = zero8();
        {
            v16bf kb[4];
#pragma unroll
            for (int j = 0; j < 4; ++j) kb[j] = loadB(Kc + j * 32, CH, lane);
#pragma unroll
            for (int j = 0; j < 4; ++j) s0 = wmma_bf16(qa[j], kb[j], s0);
#pragma unroll
            for (int j = 0; j < 4; ++j) kb[j] = loadB(Kc + 16 * CH + j * 32, CH, lane);
#pragma unroll
            for (int j = 0; j < 4; ++j) s1 = wmma_bf16(qa[j], kb[j], s1);
        }
        if (d0 + col >= DTOK) {
#pragma unroll
            for (int r = 0; r < 8; ++r) s0[r] = -3.0e38f;
        }
        if (d0 + 16 + col >= DTOK) {
#pragma unroll
            for (int r = 0; r < 8; ++r) s1[r] = -3.0e38f;
        }
        float sc[8];
#pragma unroll
        for (int r = 0; r < 8; ++r) {
            float v = fmaxf(s0[r], s1[r]);
            v = fmaxf(v, __shfl_xor(v, 1, 32));
            v = fmaxf(v, __shfl_xor(v, 2, 32));
            v = fmaxf(v, __shfl_xor(v, 4, 32));
            v = fmaxf(v, __shfl_xor(v, 8, 32));
            float nm = fmaxf(m[r], v);
            sc[r] = __expf(m[r] - nm);
            m[r] = nm;
            s0[r] = __expf(s0[r] - nm);
            s1[r] = __expf(s1[r] - nm);
            float rs = s0[r] + s1[r];
            rs += __shfl_xor(rs, 1, 32);
            rs += __shfl_xor(rs, 2, 32);
            rs += __shfl_xor(rs, 4, 32);
            rs += __shfl_xor(rs, 8, 32);
            l[r] = l[r] * sc[r] + rs;
        }
#pragma unroll
        for (int i = 0; i < 8; ++i)
#pragma unroll
            for (int r = 0; r < 8; ++r) o[i][r] *= sc[r];
        // C-layout P tile -> LDS row-major -> reload as A fragment (16x32)
#pragma unroll
        for (int r = 0; r < 8; ++r) {
            int row = rb + r;
            P[row * 32 + col] = f2bf(s0[r]);
            P[row * 32 + 16 + col] = f2bf(s1[r]);
        }
        __syncthreads();
        v16bf pa = loadA((const unsigned short*)P, 32, lane);
#pragma unroll
        for (int g = 0; g < 2; ++g) {
            v16bf vb[4];
#pragma unroll
            for (int j = 0; j < 4; ++j) {
                int i = g * 4 + j;
                vb[j] = __builtin_bit_cast(
                    v16bf, *(const v16u*)(Vc + (size_t)(i * 16 + col) * 32 + hi * 16));
            }
#pragma unroll
            for (int j = 0; j < 4; ++j) o[g * 4 + j] = wmma_bf16(pa, vb[j], o[g * 4 + j]);
        }
        __syncthreads();
    }
#pragma unroll
    for (int r = 0; r < 8; ++r) l[r] = 1.f / l[r];
#pragma unroll
    for (int i = 0; i < 8; ++i)
#pragma unroll
        for (int r = 0; r < 8; ++r)
            A0[((size_t)(n * QTOK + q0 + rb + r)) * CH + i * 16 + col] = f2bf(o[i][r] * l[r]);
}

// ---- fused epilogue: out = W_o * tanh(W_h * [x; a0; v_c] + b_h) + b_o -
__global__ __launch_bounds__(32) void out_k(const unsigned short* __restrict__ X,
                                            const unsigned short* __restrict__ A0,
                                            const unsigned short* __restrict__ VCB,
                                            const unsigned short* __restrict__ WH,
                                            const float* __restrict__ BH,
                                            const unsigned short* __restrict__ WO,
                                            const float* __restrict__ BO,
                                            float* __restrict__ OUT) {
    int lane = threadIdx.x, n = blockIdx.y, t0 = blockIdx.x * 16;
    int col = lane & 15, hi = lane >> 4, rb = hi * 8;
    __shared__ unsigned short Hl[16 * CH];   // [token][co]
    v16bf bfr[12];
#pragma unroll
    for (int j = 0; j < 4; ++j) {
        bfr[j] = loadB(X + ((size_t)(n * QTOK + t0)) * CH + j * 32, CH, lane);
        bfr[4 + j] = loadB(A0 + ((size_t)(n * QTOK + t0)) * CH + j * 32, CH, lane);
        bfr[8 + j] = __builtin_bit_cast(v16bf, *(const v16u*)(VCB + j * 32 + hi * 16));
    }
#pragma unroll
    for (int i = 0; i < 8; ++i) {
        v8f acc = zero8();
#pragma unroll
        for (int g = 0; g < 3; ++g) {
            v16bf a[4];
#pragma unroll
            for (int j = 0; j < 4; ++j)
                a[j] = loadA(WH + (size_t)(i * 16) * 384 + (g * 4 + j) * 32, 384, lane);
#pragma unroll
            for (int j = 0; j < 4; ++j) acc = wmma_bf16(a[j], bfr[g * 4 + j], acc);
        }
        v8u pk;
#pragma unroll
        for (int r = 0; r < 8; ++r) pk[r] = f2bf(tanhf(acc[r] + BH[i * 16 + rb + r]));
        *(v8u*)(Hl + (size_t)col * CH + i * 16 + rb) = pk;
    }
    __syncthreads();
#pragma unroll
    for (int i = 0; i < 8; ++i) {
        v16bf a[4], b[4];
#pragma unroll
        for (int j = 0; j < 4; ++j) {
            a[j] = loadA(WO + (size_t)(i * 16) * CH + j * 32, CH, lane);
            b[j] = __builtin_bit_cast(v16bf, *(const v16u*)(Hl + (size_t)col * CH + j * 32 + hi * 16));
        }
        v8f acc = zero8();
#pragma unroll
        for (int j = 0; j < 4; ++j) acc = wmma_bf16(a[j], b[j], acc);
#pragma unroll
        for (int r = 0; r < 8; ++r)
            OUT[((size_t)(n * CH + i * 16 + rb + r)) * QTOK + t0 + col] =
                acc[r] + BO[i * 16 + rb + r];
    }
}

// ---- host launcher ----------------------------------------------------
extern "C" void kernel_launch(void* const* d_in, const int* in_sizes, int n_in,
                              void* d_out, int out_size, void* d_ws, size_t ws_size,
                              hipStream_t stream) {
    const float* inp  = (const float*)d_in[0];
    const float* c0   = (const float*)d_in[1];
    const float* w_x  = (const float*)d_in[2];
    const float* b_x  = (const float*)d_in[3];
    const float* w_qx = (const float*)d_in[4];
    const float* w_kx = (const float*)d_in[6];
    const float* w_vx = (const float*)d_in[8];
    const float* w_vc = (const float*)d_in[9];
    const float* w_h  = (const float*)d_in[14];
    const float* b_h  = (const float*)d_in[15];
    const float* w_o  = (const float*)d_in[16];
    const float* b_o  = (const float*)d_in[17];
    float* out = (float*)d_out;

    char* ws = (char*)d_ws;
    size_t off = 0;
    auto alloc = [&](size_t bytes) -> void* {
        void* p = ws + off;
        off += (bytes + 255) & ~(size_t)255;
        return p;
    };
    unsigned short* WX  = (unsigned short*)alloc((size_t)CH * CH * 2);
    unsigned short* WQ  = (unsigned short*)alloc((size_t)CH * KCV * 2);
    unsigned short* WK  = (unsigned short*)alloc((size_t)CH * KCV * 2);
    unsigned short* WV  = (unsigned short*)alloc((size_t)CH * KCV * 2);
    unsigned short* WH  = (unsigned short*)alloc((size_t)CH * 384 * 2);
    unsigned short* WO  = (unsigned short*)alloc((size_t)CH * CH * 2);
    float*          VC  = (float*)alloc((size_t)CH * 4);
    unsigned short* VCB = (unsigned short*)alloc((size_t)CH * 2);
    unsigned short* XT  = (unsigned short*)alloc((size_t)NB * QTOK * CH * 2);
    unsigned short* Xb  = (unsigned short*)alloc((size_t)NB * QTOK * CH * 2);
    unsigned short* Qb  = (unsigned short*)alloc((size_t)NB * QTOK * CH * 2);
    unsigned short* Kb  = (unsigned short*)alloc((size_t)NB * DPAD * CH * 2);
    unsigned short* Vb  = (unsigned short*)alloc((size_t)NB * CH * DPAD * 2);
    unsigned short* A0b = (unsigned short*)alloc((size_t)NB * QTOK * CH * 2);
    (void)ws_size; (void)in_sizes; (void)n_in; (void)out_size;

    cast_bf16_k<<<(CH * CH + 255) / 256, 256, 0, stream>>>(WX, w_x, CH * CH);
    reorder_w_k<<<(CH * KCV + 255) / 256, 256, 0, stream>>>(WQ, w_qx);
    reorder_w_k<<<(CH * KCV + 255) / 256, 256, 0, stream>>>(WK, w_kx);
    reorder_w_k<<<(CH * KCV + 255) / 256, 256, 0, stream>>>(WV, w_vx);
    cast_bf16_k<<<(CH * 384 + 255) / 256, 256, 0, stream>>>(WH, w_h, CH * 384);
    cast_bf16_k<<<(CH * CH + 255) / 256, 256, 0, stream>>>(WO, w_o, CH * CH);
    vc_k<<<1, 128, 0, stream>>>(VC, VCB, w_vc, c0);
    transpose_inp_k<<<(NB * CH * QTOK + 255) / 256, 256, 0, stream>>>(XT, inp);

    dim3 blk(32);
    proj_x_k<<<dim3(QTOK / 16, NB), blk, 0, stream>>>(XT, WX, b_x, Xb);
    conv_q_k<<<dim3(QTOK / 16, NB), blk, 0, stream>>>(Xb, WQ, Qb);
    conv_kv_k<<<dim3(DPAD / 16, NB), blk, 0, stream>>>(Xb, WK, Kb, 0);
    conv_kv_k<<<dim3(DPAD / 16, NB), blk, 0, stream>>>(Xb, WV, Vb, 1);
    flash_k<<<dim3(QTOK / 16, NB), blk, 0, stream>>>(Qb, Kb, Vb, A0b);
    out_k<<<dim3(QTOK / 16, NB), blk, 0, stream>>>(Xb, A0b, VCB, WH, b_h, WO, b_o, out);
}
